// TensorVMSplit_58076547776681
// MI455X (gfx1250) — matrix operand order repaired and checked by reading, not compiled, not run on hardware
//
#include <hip/hip_runtime.h>
#include <hip/hip_bf16.h>
#include <math.h>

typedef __attribute__((ext_vector_type(16))) _Float16 v16h;
typedef __attribute__((ext_vector_type(8)))  float    v8f;

#define NRAYS   1024
#define NSAMP   690
#define PTOT    (NRAYS*NSAMP)          // 706560 = 5520 * 128
#define GRIDN   200
#define NCOMP   16
#define STEPF   (3.0f/199.0f*0.5f)
#define SH_STR  168                    // padded row stride (f16 elems) for the 128x160 tile

__device__ __forceinline__ float clampf(float v, float lo, float hi){ return fminf(fmaxf(v,lo),hi); }

__device__ __forceinline__ float ray_tmin(float ox,float oy,float oz,float dx,float dy,float dz){
  float vx = (dx==0.f)?1e-6f:dx;
  float vy = (dy==0.f)?1e-6f:dy;
  float vz = (dz==0.f)?1e-6f:dz;
  float m = fmaxf(fmaxf(fminf(( 1.5f-ox)/vx, (-1.5f-ox)/vx),
                        fminf(( 1.5f-oy)/vy, (-1.5f-oy)/vy)),
                        fminf(( 1.5f-oz)/vz, (-1.5f-oz)/vz));
  return clampf(m, 2.f, 6.f);
}

// Bilinear sample of a [16][200][200] plane at normalized (x,y), align_corners=True.
__device__ __forceinline__ void sample_plane16(const float* __restrict__ img, float x, float y, float* out){
  float fx = (x+1.f)*0.5f*(float)(GRIDN-1);
  float fy = (y+1.f)*0.5f*(float)(GRIDN-1);
  float x0 = floorf(fx), y0 = floorf(fy);
  float wx = fx - x0,  wy = fy - y0;
  int x0i = min(max((int)x0,   0), GRIDN-1);
  int x1i = min(max((int)x0+1, 0), GRIDN-1);
  int y0i = min(max((int)y0,   0), GRIDN-1);
  int y1i = min(max((int)y0+1, 0), GRIDN-1);
  float w00=(1.f-wx)*(1.f-wy), w01=wx*(1.f-wy), w10=(1.f-wx)*wy, w11=wx*wy;
  int o00=y0i*GRIDN+x0i, o01=y0i*GRIDN+x1i, o10=y1i*GRIDN+x0i, o11=y1i*GRIDN+x1i;
  #pragma unroll
  for (int c=0;c<NCOMP;c++){
    const float* b = img + c*(GRIDN*GRIDN);
    out[c] = b[o00]*w00 + b[o01]*w01 + b[o10]*w10 + b[o11]*w11;
  }
}
// Linear sample of a [16][200][1] line at normalized y (x component degenerate).
__device__ __forceinline__ void sample_line16(const float* __restrict__ img, float y, float* out){
  float fy = (y+1.f)*0.5f*(float)(GRIDN-1);
  float y0 = floorf(fy);
  float wy = fy - y0;
  int y0i = min(max((int)y0,   0), GRIDN-1);
  int y1i = min(max((int)y0+1, 0), GRIDN-1);
  #pragma unroll
  for (int c=0;c<NCOMP;c++){
    const float* b = img + c*GRIDN;
    out[c] = b[y0i]*(1.f-wy) + b[y1i]*wy;
  }
}

// ---- WMMA fragment loaders (contiguous-K order in A and B, wave32) ----
// A: 16x32 f16.  lane L: row = L%16; half=L/16; element s holds K=16*half+s (contiguous, matches B).
__device__ __forceinline__ v16h load_afrag(const _Float16* sh, int row0, int lane, int kbase){
  int ml = lane & 15, hf = lane >> 4;
  const _Float16* rp = sh + (row0 + ml)*SH_STR + kbase + 16*hf;
  v16h a;
  #pragma unroll
  for (int j=0;j<8;j++) a[j]   = rp[j];
  #pragma unroll
  for (int j=0;j<8;j++) a[8+j] = rp[8+j];
  return a;
}
// B: 32x16 f16 from weight buffer stored row-major [n][K] (i.e. W.T columns contiguous in K).
// lane L: col = L%16; half=L/16; VGPR j holds K = 16*half + {2j,2j+1} -> 16 contiguous halves.
__device__ __forceinline__ v16h load_bfrag(const _Float16* __restrict__ wb, int Kst, int ncol0, int lane, int kbase){
  int nl = lane & 15, hf = lane >> 4;
  const _Float16* wp = wb + (size_t)(ncol0 + nl)*Kst + kbase + 16*hf;
  v16h b;
  #pragma unroll
  for (int j=0;j<16;j++) b[j] = wp[j];
  return b;
}

// ---- K0: convert fp32 weights to fp16 in WMMA-friendly [n][K] layout, with zero padding ----
__global__ void k_convert(const float* __restrict__ W1, const float* __restrict__ W2,
                          const float* __restrict__ W3,
                          _Float16* __restrict__ w1h, _Float16* __restrict__ w2h,
                          _Float16* __restrict__ w3h){
  int i = blockIdx.x*blockDim.x + threadIdx.x;
  if (i < 128*160){
    int n = i/160, k = i - n*160;
    w1h[i] = (_Float16)((k<150) ? W1[n*150+k] : 0.f);
  } else if (i < 128*160 + 128*128){
    int j = i - 128*160;
    w2h[j] = (_Float16)W2[j];                       // [n][k] == W2 row-major
  } else if (i < 128*160 + 128*128 + 16*128){
    int j = i - 128*160 - 128*128;
    int n = j/128, k = j - n*128;
    w3h[j] = (_Float16)((n<3) ? W3[n*128+k] : 0.f); // pad N 3->16 with zeros
  }
}

// ---- K1: fused featurize (grid sampling + PE) + 3-layer MLP via WMMA ----
__global__ __launch_bounds__(256) void k_fused(
  const float* __restrict__ rays,
  const float* __restrict__ planes_d, const float* __restrict__ lines_d,
  const float* __restrict__ planes_c, const float* __restrict__ lines_c,
  const float* __restrict__ basisW,
  const float* __restrict__ b1, const float* __restrict__ b2, const float* __restrict__ b3,
  const _Float16* __restrict__ w1h, const _Float16* __restrict__ w2h, const _Float16* __restrict__ w3h,
  float* __restrict__ sigma_ws, float* __restrict__ rgb_ws)
{
  __shared__ _Float16 sh[128*SH_STR];   // 43 KB: mlp_in tile, then reused for h1/h2
  const int t  = threadIdx.x;
  const int p0 = blockIdx.x*128;
  const int lp = t & 127;
  const int p  = p0 + lp;
  const int r  = p / NSAMP;
  const int s  = p - r*NSAMP;

  float ox=rays[r*6+0], oy=rays[r*6+1], oz=rays[r*6+2];
  float dx=rays[r*6+3], dy=rays[r*6+4], dz=rays[r*6+5];
  float tmin = ray_tmin(ox,oy,oz,dx,dy,dz);
  float z  = tmin + STEPF*(float)s;
  float px = ox+dx*z, py = oy+dy*z, pz = oz+dz*z;
  bool valid = (fabsf(px)<=1.5f) && (fabsf(py)<=1.5f) && (fabsf(pz)<=1.5f);
  float xn = px*(2.f/3.f), yn = py*(2.f/3.f), zn = pz*(2.f/3.f);

  if (t >= 128){
    // ---- density half: sigma_feat -> softplus -> workspace ----
    float pl[NCOMP], ln[NCOMP];
    float sf = 0.f;
    sample_plane16(planes_d + 0*NCOMP*GRIDN*GRIDN, xn, yn, pl);   // mat (0,1)
    sample_line16 (lines_d  + 0*NCOMP*GRIDN,       zn, ln);       // vec 2
    #pragma unroll
    for (int c=0;c<NCOMP;c++) sf += pl[c]*ln[c];
    sample_plane16(planes_d + 1*NCOMP*GRIDN*GRIDN, xn, zn, pl);   // mat (0,2)
    sample_line16 (lines_d  + 1*NCOMP*GRIDN,       yn, ln);       // vec 1
    #pragma unroll
    for (int c=0;c<NCOMP;c++) sf += pl[c]*ln[c];
    sample_plane16(planes_d + 2*NCOMP*GRIDN*GRIDN, yn, zn, pl);   // mat (1,2)
    sample_line16 (lines_d  + 2*NCOMP*GRIDN,       xn, ln);       // vec 0
    #pragma unroll
    for (int c=0;c<NCOMP;c++) sf += pl[c]*ln[c];
    float sg = 0.f;
    if (valid){
      float xsf = sf - 10.f;                                      // DENSE_SHIFT
      sg = (xsf > 20.f) ? xsf : log1pf(__expf(xsf));              // softplus
    }
    sigma_ws[p] = sg;
  } else {
    // ---- appearance half: 48 color feats -> basis -> mlp_in row (fp16, LDS) ----
    float cf[48];
    float pl[NCOMP], ln[NCOMP];
    sample_plane16(planes_c + 0*NCOMP*GRIDN*GRIDN, xn, yn, pl);
    sample_line16 (lines_c  + 0*NCOMP*GRIDN,       zn, ln);
    #pragma unroll
    for (int c=0;c<NCOMP;c++) cf[c]      = pl[c]*ln[c];
    sample_plane16(planes_c + 1*NCOMP*GRIDN*GRIDN, xn, zn, pl);
    sample_line16 (lines_c  + 1*NCOMP*GRIDN,       yn, ln);
    #pragma unroll
    for (int c=0;c<NCOMP;c++) cf[16+c]   = pl[c]*ln[c];
    sample_plane16(planes_c + 2*NCOMP*GRIDN*GRIDN, yn, zn, pl);
    sample_line16 (lines_c  + 2*NCOMP*GRIDN,       xn, ln);
    #pragma unroll
    for (int c=0;c<NCOMP;c++) cf[32+c]   = pl[c]*ln[c];

    _Float16* rp = &sh[lp*SH_STR];
    rp[27] = (_Float16)dx; rp[28] = (_Float16)dy; rp[29] = (_Float16)dz;
    for (int o=0;o<27;o++){
      float f = 0.f;
      const float* bw = basisW + o*48;
      #pragma unroll
      for (int j=0;j<48;j++) f += cf[j]*bw[j];
      rp[o] = (_Float16)f;
      float s1,c1,s2,c2;
      __sincosf(f, &s1, &c1);
      __sincosf(2.f*f, &s2, &c2);
      rp[30+2*o]   = (_Float16)s1; rp[30+2*o+1] = (_Float16)s2;
      rp[84+2*o]   = (_Float16)c1; rp[84+2*o+1] = (_Float16)c2;
    }
    float rd3[3] = {dx,dy,dz};
    #pragma unroll
    for (int c=0;c<3;c++){
      float s1,c1,s2,c2;
      __sincosf(rd3[c], &s1, &c1);
      __sincosf(2.f*rd3[c], &s2, &c2);
      rp[138+2*c] = (_Float16)s1; rp[138+2*c+1] = (_Float16)s2;
      rp[144+2*c] = (_Float16)c1; rp[144+2*c+1] = (_Float16)c2;
    }
    #pragma unroll
    for (int k=150;k<160;k++) rp[k] = (_Float16)0.f;
  }
  __syncthreads();

  // ---- MLP via WMMA: each wave owns a 16-row stripe ----
  const int wave = t >> 5, lane = t & 31;
  const int row0 = wave*16;
  const int nl = lane & 15, hf = lane >> 4;

  v8f acc[8];
  // Layer 1: [128x160] x [160x128], bias in C init
  #pragma unroll
  for (int nt=0;nt<8;nt++){
    float bv = b1[nt*16+nl];
    #pragma unroll
    for (int j=0;j<8;j++) acc[nt][j] = bv;
  }
  #pragma unroll
  for (int ks=0;ks<5;ks++){
    v16h a = load_afrag(sh, row0, lane, ks*32);
    #pragma unroll
    for (int nt=0;nt<8;nt++){
      v16h b = load_bfrag(w1h, 160, nt*16, lane, ks*32);
      acc[nt] = __builtin_amdgcn_wmma_f32_16x16x32_f16(false, a, false, b, (short)0, acc[nt], false, false);
    }
  }
  #pragma unroll
  for (int nt=0;nt<8;nt++){
    #pragma unroll
    for (int j=0;j<8;j++){
      float v = fmaxf(acc[nt][j], 0.f);                       // ReLU
      sh[(row0 + j + 8*hf)*SH_STR + nt*16 + nl] = (_Float16)v;
    }
  }

  // Layer 2: [128x128] x [128x128]
  #pragma unroll
  for (int nt=0;nt<8;nt++){
    float bv = b2[nt*16+nl];
    #pragma unroll
    for (int j=0;j<8;j++) acc[nt][j] = bv;
  }
  #pragma unroll
  for (int ks=0;ks<4;ks++){
    v16h a = load_afrag(sh, row0, lane, ks*32);
    #pragma unroll
    for (int nt=0;nt<8;nt++){
      v16h b = load_bfrag(w2h, 128, nt*16, lane, ks*32);
      acc[nt] = __builtin_amdgcn_wmma_f32_16x16x32_f16(false, a, false, b, (short)0, acc[nt], false, false);
    }
  }
  #pragma unroll
  for (int nt=0;nt<8;nt++){
    #pragma unroll
    for (int j=0;j<8;j++){
      float v = fmaxf(acc[nt][j], 0.f);
      sh[(row0 + j + 8*hf)*SH_STR + nt*16 + nl] = (_Float16)v;
    }
  }

  // Layer 3: [128x128] x [128x16] (3 valid cols, zero-padded)
  v8f acc3;
  {
    float bv = (nl<3) ? b3[nl] : 0.f;
    #pragma unroll
    for (int j=0;j<8;j++) acc3[j] = bv;
  }
  #pragma unroll
  for (int ks=0;ks<4;ks++){
    v16h a = load_afrag(sh, row0, lane, ks*32);
    v16h b = load_bfrag(w3h, 128, 0, lane, ks*32);
    acc3 = __builtin_amdgcn_wmma_f32_16x16x32_f16(false, a, false, b, (short)0, acc3, false, false);
  }
  if (nl < 3){
    #pragma unroll
    for (int j=0;j<8;j++){
      int m  = j + 8*hf;
      int pp = p0 + row0 + m;
      float v = acc3[j];
      rgb_ws[pp*3 + nl] = 1.f/(1.f + __expf(-v));             // sigmoid
    }
  }
}

// ---- K2: one wave per ray; parallel transmittance scan + composite ----
__global__ __launch_bounds__(256) void k_composite(
  const float* __restrict__ rays, const float* __restrict__ sigma_ws,
  const float* __restrict__ rgb_ws, float* __restrict__ out)
{
  const int lane = threadIdx.x & 31;
  const int ray  = blockIdx.x*8 + (threadIdx.x >> 5);
  float ox=rays[ray*6+0], oy=rays[ray*6+1], oz=rays[ray*6+2];
  float dx=rays[ray*6+3], dy=rays[ray*6+4], dz=rays[ray*6+5];
  float tmin = ray_tmin(ox,oy,oz,dx,dy,dz);

  float T = 1.f;
  float accR=0.f, accG=0.f, accB=0.f, wsum=0.f, wzsum=0.f;

  for (int s0=0; s0<NSAMP; s0+=32){
    int s = s0 + lane;
    bool in = s < NSAMP;
    float sg    = in ? sigma_ws[ray*NSAMP + s] : 0.f;
    float dist  = (s < NSAMP-1) ? STEPF : 0.f;                // last dist = 0
    float alpha = in ? (1.f - __expf(-sg*dist*25.f)) : 0.f;
    float om    = in ? (1.f - alpha + 1e-10f) : 1.f;
    // inclusive prefix product of om across the wave
    float scan = om;
    #pragma unroll
    for (int off=1; off<32; off<<=1){
      float v = __shfl_up(scan, off, 32);
      if (lane >= off) scan *= v;
    }
    float excl = __shfl_up(scan, 1, 32);
    if (lane == 0) excl = 1.f;
    float w = alpha * T * excl;
    if (in){
      float zv = tmin + STEPF*(float)s;
      wsum  += w;
      wzsum += w*zv;
      if (w > 1e-4f){                                         // W_THRESH mask
        int b = (ray*NSAMP + s)*3;
        accR += w*rgb_ws[b+0];
        accG += w*rgb_ws[b+1];
        accB += w*rgb_ws[b+2];
      }
    }
    T *= __shfl(scan, 31, 32);
  }
  #pragma unroll
  for (int off=16; off>0; off>>=1){
    accR  += __shfl_down(accR , off, 32);
    accG  += __shfl_down(accG , off, 32);
    accB  += __shfl_down(accB , off, 32);
    wsum  += __shfl_down(wsum , off, 32);
    wzsum += __shfl_down(wzsum, off, 32);
  }
  if (lane == 0){
    float bg = 1.f - wsum;
    out[ray*3+0] = clampf(accR + bg, 0.f, 1.f);
    out[ray*3+1] = clampf(accG + bg, 0.f, 1.f);
    out[ray*3+2] = clampf(accB + bg, 0.f, 1.f);
    out[3*NRAYS + ray] = wzsum + bg*dz;                       // depth
  }
}

extern "C" void kernel_launch(void* const* d_in, const int* in_sizes, int n_in,
                              void* d_out, int out_size, void* d_ws, size_t ws_size,
                              hipStream_t stream) {
  (void)in_sizes; (void)n_in; (void)out_size; (void)ws_size;
  const float* rays     = (const float*)d_in[0];
  const float* planes_d = (const float*)d_in[1];
  const float* lines_d  = (const float*)d_in[2];
  const float* planes_c = (const float*)d_in[3];
  const float* lines_c  = (const float*)d_in[4];
  const float* basisW   = (const float*)d_in[5];
  const float* W1 = (const float*)d_in[6];
  const float* b1 = (const float*)d_in[7];
  const float* W2 = (const float*)d_in[8];
  const float* b2 = (const float*)d_in[9];
  const float* W3 = (const float*)d_in[10];
  const float* b3 = (const float*)d_in[11];

  char* ws = (char*)d_ws;
  size_t off = 0;
  float* sigma_ws = (float*)(ws + off); off += (size_t)PTOT*4;    off = (off+255) & ~(size_t)255;
  float* rgb_ws   = (float*)(ws + off); off += (size_t)PTOT*3*4;  off = (off+255) & ~(size_t)255;
  _Float16* w1h   = (_Float16*)(ws + off); off += 128*160*2;      off = (off+255) & ~(size_t)255;
  _Float16* w2h   = (_Float16*)(ws + off); off += 128*128*2;      off = (off+255) & ~(size_t)255;
  _Float16* w3h   = (_Float16*)(ws + off); off += 16*128*2;

  int convTotal = 128*160 + 128*128 + 16*128;
  k_convert<<<(convTotal+255)/256, 256, 0, stream>>>(W1, W2, W3, w1h, w2h, w3h);
  k_fused<<<PTOT/128, 256, 0, stream>>>(rays, planes_d, lines_d, planes_c, lines_c,
                                        basisW, b1, b2, b3, w1h, w2h, w3h,
                                        sigma_ws, rgb_ws);
  k_composite<<<NRAYS/8, 256, 0, stream>>>(rays, sigma_ws, rgb_ws, (float*)d_out);
}
